// RNN_26895085207931
// MI455X (gfx1250) — matrix operand order repaired and checked by reading, not compiled, not run on hardware
//
#include <hip/hip_runtime.h>
#include <cstdint>
#include <cstddef>

// ---------------------------------------------------------------------------
// CDNA5 (gfx1250) bf16 WMMA RNN: pipelined WMMA GEMMs + persistent recurrence.
// ---------------------------------------------------------------------------

typedef __attribute__((ext_vector_type(16))) __bf16 v16bf;
typedef __attribute__((ext_vector_type(8)))  float  v8f;

union FragU {
    unsigned short us[16];
    uint4          q[2];
    v16bf          v;
};

__device__ __forceinline__ unsigned short f2bf(float f) {
    union { float f; unsigned int u; } c; c.f = f;
    unsigned int u = c.u + 0x7FFFu + ((c.u >> 16) & 1u);   // round-to-nearest-even
    return (unsigned short)(u >> 16);
}
__device__ __forceinline__ float bf2f(unsigned short h) {
    union { unsigned int u; float f; } c; c.u = ((unsigned int)h) << 16;
    return c.f;
}

// ---------------------------------------------------------------------------
// Elementwise f32 -> bf16 conversion.
// ---------------------------------------------------------------------------
__global__ __launch_bounds__(256)
void f32_to_bf16_kernel(const float* __restrict__ in,
                        unsigned short* __restrict__ out, size_t n) {
    size_t i = (size_t)blockIdx.x * blockDim.x + threadIdx.x;
    if (i < n) out[i] = f2bf(in[i]);
}

__global__ __launch_bounds__(64)
void zero_u32_kernel(unsigned* __restrict__ p, int n) {
    int i = blockIdx.x * blockDim.x + threadIdx.x;
    if (i < n) p[i] = 0u;
}

// ---------------------------------------------------------------------------
// Pack a row-major fp32 weight W[G][K] (used as B[k][n] = W[n][k]) into
// per-lane B-fragment order used here (contiguous K, same as A) for v_wmma_f32_16x16x32_bf16:
//   B is 32x16 (KxN); lanes 0-15: N=lane, VGPR r holds K=2r,2r+1
//                     lanes 16-31: N=lane-16, VGPR r holds K=16+2r,16+2r+1
// => lane packs 16 consecutive K values starting at hi*16, fixed N row of W.
// Packed layout: tile (nt,kt) -> base ((nt*ktiles + kt)*32 + lane)*16 bf16.
// ---------------------------------------------------------------------------
__global__ __launch_bounds__(256)
void pack_weight_kernel(const float* __restrict__ W,
                        unsigned short* __restrict__ P, int G, int K) {
    int tid    = blockIdx.x * blockDim.x + threadIdx.x;
    int ktiles = K >> 5;
    int total  = (G >> 4) * ktiles * 32;
    if (tid >= total) return;
    int lane = tid & 31;
    int tIdx = tid >> 5;
    int nt   = tIdx / ktiles;
    int kt   = tIdx - nt * ktiles;
    int n    = lane & 15;
    int hi   = lane >> 4;
    const float* src = W + (size_t)(nt * 16 + n) * K + kt * 32 + hi * 16;
    unsigned short* dst = P + (size_t)tid * 16;
#pragma unroll
    for (int j = 0; j < 16; ++j) dst[j] = f2bf(src[j]);
}

// ---------------------------------------------------------------------------
// Tiled bf16 WMMA GEMM:  C[M][N] = A[M][K] @ Bpackedᵀ + bias
// Each wave computes a 16x64 strip: 4 accumulators, K-loop in steps of 32,
// 2-deep software pipeline (prefetch kt+2/kt+3 while computing kt/kt+1).
// ---------------------------------------------------------------------------
__global__ __launch_bounds__(256)
void gemm_bf16_kernel(const unsigned short* __restrict__ A,
                      const unsigned short* __restrict__ Bp,
                      const float* __restrict__ bias,
                      unsigned short* __restrict__ outBf,
                      float* __restrict__ outF,
                      int M, int N, int K) {
    const int lane = threadIdx.x & 31;
    const int wid  = (int)((blockIdx.x * blockDim.x + threadIdx.x) >> 5);
    const int NC   = N >> 6;                 // 64-col chunks
    const int mt   = wid / NC;
    const int nc   = wid - mt * NC;
    if (mt * 16 >= M) return;

    const int ktiles = K >> 5;               // even (16 or 32)
    const int mrow   = mt * 16 + (lane & 15);
    const int aoff   = (lane >> 4) ? 16 : 0; // A layout: hi lanes hold K=16..31
    const unsigned short* abase = A + (size_t)mrow * K + aoff;

    const size_t bNtStride = (size_t)ktiles * 32 * 16;  // bf16 elems per nt step
    const unsigned short* bbase =
        Bp + (size_t)(nc * 4) * bNtStride + ((size_t)lane * 16);

    v8f acc[4] = {};

    auto loadA = [&](FragU& a, int kt) {
        const unsigned short* p = abase + (size_t)kt * 32;
        a.q[0] = *(const uint4*)(p);
        a.q[1] = *(const uint4*)(p + 8);
    };
    auto loadB = [&](FragU* b, int kt) {
        const unsigned short* p = bbase + (size_t)kt * 32 * 16;
#pragma unroll
        for (int t = 0; t < 4; ++t) {
            b[t].q[0] = *(const uint4*)(p);
            b[t].q[1] = *(const uint4*)(p + 8);
            p += bNtStride;
        }
    };
    auto compute = [&](const FragU& a, const FragU* b) {
#pragma unroll
        for (int t = 0; t < 4; ++t)
            acc[t] = __builtin_amdgcn_wmma_f32_16x16x32_bf16(
                false, a.v, false, b[t].v, (short)0, acc[t], false, false);
    };

    FragU a0, a1, b0[4], b1[4];
    loadA(a0, 0); loadB(b0, 0);
    loadA(a1, 1); loadB(b1, 1);
    for (int kt = 0; kt + 2 < ktiles; kt += 2) {
        compute(a0, b0);
        loadA(a0, kt + 2); loadB(b0, kt + 2);   // prefetch while b1-set computes
        compute(a1, b1);
        loadA(a1, kt + 3); loadB(b1, kt + 3);
    }
    compute(a0, b0);
    compute(a1, b1);

    const int hi8 = (lane >> 4) * 8;
#pragma unroll
    for (int t = 0; t < 4; ++t) {
        const int col = nc * 64 + t * 16 + (lane & 15);
        const float bv = bias ? bias[col] : 0.0f;
#pragma unroll
        for (int r = 0; r < 8; ++r) {
            const int row = mt * 16 + hi8 + r;
            const float v = acc[t][r] + bv;
            if (outF) outF[(size_t)row * N + col] = v;
            else      outBf[(size_t)row * N + col] = f2bf(v);
        }
    }
}

// ---------------------------------------------------------------------------
// Persistent recurrence kernel: runs ALL Tsteps timesteps of one layer in a
// single launch.  Per step:  Buf[t] = tanh(Buf[t] + h_{t-1} @ W_hhᵀ + b_hh)
// (in place; h_{t-1} = Buf[t-1], or h0 for t==0).
// Exactly (Bsz/16)*(H/16) waves = 32 blocks x 256 threads -> all co-resident
// on MI455X, so an atomic count/generation grid barrier is deadlock-free.
// __threadfence() provides agent-scope release before arrival and the
// acquire/invalidate after release (stale WGP$ lines from step t-1 fills).
// ---------------------------------------------------------------------------
__global__ __launch_bounds__(256)
void rnn_persistent_kernel(const unsigned short* __restrict__ h0,
                           const unsigned short* __restrict__ Whhp,
                           const float* __restrict__ bhh,
                           unsigned short* __restrict__ Buf,   // [Tsteps][Bsz][H]
                           unsigned* __restrict__ bar,          // bar[0]=cnt, bar[16]=gen
                           int Tsteps, int H, int Bsz) {
    const int lane   = threadIdx.x & 31;
    const int wid    = (int)((blockIdx.x * blockDim.x + threadIdx.x) >> 5);
    const int ntiles = H >> 4;
    const int mt     = wid / ntiles;     // grid sized exactly: no guard (syncthreads!)
    const int nt     = wid - mt * ntiles;
    const unsigned nb = gridDim.x;

    const int ktiles = H >> 5;                       // 32
    const int mrow   = mt * 16 + (lane & 15);
    const int aoff   = (lane >> 4) ? 16 : 0;
    const size_t arowoff = (size_t)mrow * H + aoff;
    const unsigned short* bbase = Whhp + ((size_t)nt * ktiles * 32 + lane) * 16;

    const int   col  = nt * 16 + (lane & 15);
    const float bv   = bhh[col];                     // invariant across steps
    const int   hi8  = (lane >> 4) * 8;
    const size_t slab = (size_t)Bsz * H;

    for (int t = 0; t < Tsteps; ++t) {
        const unsigned short* abase =
            ((t == 0) ? h0 : (Buf + (size_t)(t - 1) * slab)) + arowoff;
        unsigned short* xio = Buf + (size_t)t * slab;

        auto loadAB = [&](FragU& a, FragU& b, int kt) {
            const unsigned short* pa = abase + (size_t)kt * 32;
            a.q[0] = *(const uint4*)(pa);
            a.q[1] = *(const uint4*)(pa + 8);
            const unsigned short* pb = bbase + (size_t)kt * 32 * 16;
            b.q[0] = *(const uint4*)(pb);
            b.q[1] = *(const uint4*)(pb + 8);
        };

        v8f acc = {};
        FragU a0, b0, a1, b1;
        loadAB(a0, b0, 0);
        loadAB(a1, b1, 1);
        for (int kt = 0; kt + 2 < ktiles; kt += 2) {
            acc = __builtin_amdgcn_wmma_f32_16x16x32_bf16(
                false, a0.v, false, b0.v, (short)0, acc, false, false);
            loadAB(a0, b0, kt + 2);
            acc = __builtin_amdgcn_wmma_f32_16x16x32_bf16(
                false, a1.v, false, b1.v, (short)0, acc, false, false);
            loadAB(a1, b1, kt + 3);
        }
        acc = __builtin_amdgcn_wmma_f32_16x16x32_bf16(
            false, a0.v, false, b0.v, (short)0, acc, false, false);
        acc = __builtin_amdgcn_wmma_f32_16x16x32_bf16(
            false, a1.v, false, b1.v, (short)0, acc, false, false);

#pragma unroll
        for (int r = 0; r < 8; ++r) {
            const size_t idx = (size_t)(mt * 16 + hi8 + r) * H + col;
            const float v = tanhf(acc[r] + bf2f(xio[idx]) + bv);
            xio[idx] = f2bf(v);
        }

        if (t + 1 < Tsteps) {
            // ---- grid-wide barrier (count + generation) ----
            __threadfence();                 // release this block's stores
            __syncthreads();
            if (threadIdx.x == 0) {
                unsigned g = __hip_atomic_load(&bar[16], __ATOMIC_RELAXED,
                                               __HIP_MEMORY_SCOPE_AGENT);
                unsigned prev = __hip_atomic_fetch_add(&bar[0], 1u, __ATOMIC_ACQ_REL,
                                                       __HIP_MEMORY_SCOPE_AGENT);
                if (prev == nb - 1u) {
                    __hip_atomic_store(&bar[0], 0u, __ATOMIC_RELAXED,
                                       __HIP_MEMORY_SCOPE_AGENT);
                    __hip_atomic_fetch_add(&bar[16], 1u, __ATOMIC_RELEASE,
                                           __HIP_MEMORY_SCOPE_AGENT);
                } else {
                    while (__hip_atomic_load(&bar[16], __ATOMIC_ACQUIRE,
                                             __HIP_MEMORY_SCOPE_AGENT) == g)
                        __builtin_amdgcn_s_sleep(1);
                }
            }
            __syncthreads();
            __threadfence();                 // acquire: drop stale WGP$ lines
        }
    }
}

// ---------------------------------------------------------------------------
// Host orchestration.
// ---------------------------------------------------------------------------
extern "C" void kernel_launch(void* const* d_in, const int* in_sizes, int n_in,
                              void* d_out, int out_size, void* d_ws, size_t ws_size,
                              hipStream_t stream) {
    (void)in_sizes; (void)n_in; (void)out_size; (void)ws_size;
    constexpr int T = 512, Bb = 64, I = 512, H = 1024;

    const float* x      = (const float*)d_in[0];
    const float* hidden = (const float*)d_in[1];
    const float* W_ih0  = (const float*)d_in[2];
    const float* W_hh0  = (const float*)d_in[3];
    const float* b_ih0  = (const float*)d_in[4];
    const float* b_hh0  = (const float*)d_in[5];
    const float* W_ih1  = (const float*)d_in[6];
    const float* W_hh1  = (const float*)d_in[7];
    const float* b_ih1  = (const float*)d_in[8];
    const float* b_hh1  = (const float*)d_in[9];
    const float* W_fc   = (const float*)d_in[10];
    const float* b_fc   = (const float*)d_in[11];
    float* out = (float*)d_out;

    // Workspace carve-out (256B aligned chunks). Total ~168 MB.
    char*  ws  = (char*)d_ws;
    size_t off = 0;
    auto carve = [&](size_t bytes) -> void* {
        void* p = ws + off;
        off += (bytes + 255) & ~(size_t)255;
        return p;
    };
    unsigned*       bar   = (unsigned*)carve(128);
    unsigned short* Wih0p = (unsigned short*)carve((size_t)H * I * 2);
    unsigned short* Whh0p = (unsigned short*)carve((size_t)H * H * 2);
    unsigned short* Wih1p = (unsigned short*)carve((size_t)H * H * 2);
    unsigned short* Whh1p = (unsigned short*)carve((size_t)H * H * 2);
    unsigned short* Wfcp  = (unsigned short*)carve((size_t)I * H * 2);
    unsigned short* hbf   = (unsigned short*)carve((size_t)2 * Bb * H * 2);
    unsigned short* xbf   = (unsigned short*)carve((size_t)T * Bb * I * 2);
    unsigned short* BufA  = (unsigned short*)carve((size_t)T * Bb * H * 2);
    unsigned short* BufB  = (unsigned short*)carve((size_t)T * Bb * H * 2);

    const int M = T * Bb;  // 32768 GEMM rows

    // 1) Convert activations/initial hidden to bf16.
    {
        size_t n = (size_t)T * Bb * I;
        f32_to_bf16_kernel<<<(unsigned)((n + 255) / 256), 256, 0, stream>>>(x, xbf, n);
        size_t nh = (size_t)2 * Bb * H;
        f32_to_bf16_kernel<<<(unsigned)((nh + 255) / 256), 256, 0, stream>>>(hidden, hbf, nh);
    }

    // 2) Pack weights into WMMA B-fragment order.
    auto pack = [&](const float* W, unsigned short* P, int G, int K) {
        int threads = (G / 16) * (K / 32) * 32;
        pack_weight_kernel<<<(threads + 255) / 256, 256, 0, stream>>>(W, P, G, K);
    };
    pack(W_ih0, Wih0p, H, I);
    pack(W_hh0, Whh0p, H, H);
    pack(W_ih1, Wih1p, H, H);
    pack(W_hh1, Whh1p, H, H);
    pack(W_fc,  Wfcp,  I, H);

    auto gemm = [&](const unsigned short* A, const unsigned short* Bp,
                    const float* bias, unsigned short* oBf, float* oF,
                    int N, int K) {
        int waves  = (M / 16) * (N / 64);
        int blocks = waves * 32 / 256;
        gemm_bf16_kernel<<<blocks, 256, 0, stream>>>(A, Bp, bias, oBf, oF, M, N, K);
    };

    const int stepBlocks = ((Bb / 16) * (H / 16) * 32) / 256;  // 32 blocks, exact grid

    // 3) Layer 0: xproj = x @ W_ih0ᵀ + b_ih0, then persistent 512-step recurrence.
    gemm(xbf, Wih0p, b_ih0, BufA, nullptr, H, I);
    zero_u32_kernel<<<1, 64, 0, stream>>>(bar, 32);
    rnn_persistent_kernel<<<stepBlocks, 256, 0, stream>>>(
        hbf, Whh0p, b_hh0, BufA, bar, T, H, Bb);

    // 4) Layer 1: xproj = out0 @ W_ih1ᵀ + b_ih1, then persistent recurrence.
    gemm(BufA, Wih1p, b_ih1, BufB, nullptr, H, H);
    zero_u32_kernel<<<1, 64, 0, stream>>>(bar, 32);
    rnn_persistent_kernel<<<stepBlocks, 256, 0, stream>>>(
        hbf + (size_t)Bb * H, Whh1p, b_hh1, BufB, bar, T, H, Bb);

    // 5) Final Linear: out = out1 @ W_fcᵀ + b_fc  (fp32 output).
    gemm(BufB, Wfcp, b_fc, nullptr, out, I, H);
}
